// STGumbel_AR_Tree_8873402433918
// MI455X (gfx1250) — matrix operand order repaired and checked by reading, not compiled, not run on hardware
//
#include <hip/hip_runtime.h>
#include <hip/hip_bf16.h>

// ---------------- problem constants ----------------
#define Bb 8
#define Tt 64
#define Dd 512
#define Hh 1024
#define LEAF_BASE 100000

typedef __attribute__((ext_vector_type(16))) __bf16 v16bf;
typedef __attribute__((ext_vector_type(8)))  float  v8f;

union Frag { unsigned int u[8]; v16bf v; };

__device__ __forceinline__ unsigned int pack2bf(float a, float b) {
    union { __bf16 h[2]; unsigned int u; } p;
    p.h[0] = (__bf16)a; p.h[1] = (__bf16)b;
    return p.u;
}
__device__ __forceinline__ float sigf(float x) { return 1.f / (1.f + __expf(-x)); }

// ---------------- generic 16x16 WMMA tile (bf16, f32 acc) ----------------
// Apk:  A packed pairs, dword index = kpair*16 + m   (m = padded batch row)
// Bsw:  pre-swizzled B base for this ntile; dword index = kt*ktStride + r*32 + lane
__device__ __forceinline__ v8f gemm_tile(const unsigned int* __restrict__ Apk,
                                         const unsigned int* __restrict__ Bsw,
                                         int kTiles, int ktStride, int lane) {
    v8f acc = {};
    const int half = lane >> 4, ml = lane & 15;
    for (int kt = 0; kt < kTiles; ++kt) {
        Frag A, Bf;
#pragma unroll
        for (int r = 0; r < 8; ++r) {
            // A fragment: lane l, reg r holds K = 2*r + 16*half (+1)
            int kp = kt * 16 + (half << 3) + r;
            A.u[r]  = Apk[kp * 16 + ml];
            Bf.u[r] = Bsw[kt * ktStride + r * 32 + lane];
        }
        acc = __builtin_amdgcn_wmma_f32_16x16x32_bf16(
            /*neg_a=*/false, A.v, /*neg_b=*/false, Bf.v,
            /*c_mod=*/(short)0, acc, /*reuse_a=*/false, /*reuse_b=*/false);
    }
    return acc;
}

// ---------------- prep: swizzle W[N][K] f32 -> bf16 B fragments ----------------
// Bs dword index = ((kt*NT + nt)*8 + r)*32 + lane
// element: n = nt*16 + (lane&15); k = kt*32 + 16*(lane>>4) + 2r (+1);  B[k][n] = W[n][k]
__global__ void swizzleB(const float* __restrict__ W, unsigned int* __restrict__ Bs,
                         int N, int K) {
    int idx = blockIdx.x * 256 + threadIdx.x;
    int total = (N / 16) * (K / 32) * 256;
    if (idx >= total) return;
    int lane = idx & 31, r = (idx >> 5) & 7, tile = idx >> 8;
    int NT = N / 16;
    int nt = tile % NT, kt = tile / NT;
    int n = nt * 16 + (lane & 15);
    int k = kt * 32 + ((lane >> 4) << 4) + 2 * r;
    Bs[idx] = pack2bf(W[n * K + k], W[n * K + k + 1]);
}

// ---------------- prep: pack sentence_embedding into A pairs ----------------
// Apk idx = (mt*(D/2) + kp)*16 + ml ; row m = b*T + t (flat, matches (B,T,D))
__global__ void packA_emb(const float* __restrict__ emb, unsigned int* __restrict__ Apk) {
    int idx = blockIdx.x * 256 + threadIdx.x;
    if (idx >= 32 * (Dd / 2) * 16) return;
    int ml = idx & 15, kp = (idx >> 4) & (Dd / 2 - 1), mt = idx >> 12;
    const float* row = emb + (mt * 16 + ml) * Dd;
    Apk[idx] = pack2bf(row[2 * kp], row[2 * kp + 1]);
}

__global__ void zero_u32(unsigned int* p, int n) {
    int i = blockIdx.x * 256 + threadIdx.x;
    if (i < n) p[i] = 0u;
}

// ---------------- x-gate precompute: (B*T x D) @ W_ih^T + biases ----------------
// grid 2048 x 128; wave = one 16x16 tile. xg layout: [t][n(0..4095)][b]
__global__ void gemm_xg(const unsigned int* __restrict__ Apk,
                        const unsigned int* __restrict__ Bsw,
                        const float* __restrict__ b_ih, const float* __restrict__ b_hh,
                        float* __restrict__ xg) {
    int lane = threadIdx.x & 31, w = threadIdx.x >> 5;
    int tile = blockIdx.x * 4 + w;
    int mt = tile >> 8, nt = tile & 255;
    v8f acc = gemm_tile(Apk + mt * (Dd / 2) * 16, Bsw + nt * 256, Dd / 32, 256 * 256, lane);
    int half = lane >> 4, n = nt * 16 + (lane & 15);
    float bias = b_ih[n] + b_hh[n];
#pragma unroll
    for (int r = 0; r < 8; ++r) {
        int m = mt * 16 + r + 8 * half;          // m = b*T + t
        int b = m >> 6, t = m & 63;
        xg[(t * 4096 + n) * Bb + b] = acc[r] + bias;
    }
}

// ---------------- one LSTM recurrence step ----------------
// grid 64 x 128 (4 waves). Block owns hidden chunk j0..j0+15; wave g = gate g.
__global__ void lstm_step(int t, const unsigned int* __restrict__ Ain,
                          unsigned int* __restrict__ Aout,
                          const unsigned int* __restrict__ Whh,
                          const float* __restrict__ xg,
                          float* __restrict__ hs, float* __restrict__ cs) {
    __shared__ float gl[4][16][16];
    __shared__ float hsh[Bb][16];
    int lane = threadIdx.x & 31, w = threadIdx.x >> 5;
    int j0 = blockIdx.x * 16;
    int ntile = w * 64 + blockIdx.x;                 // column tile in 4096
    v8f acc = gemm_tile(Ain, Whh + ntile * 256, Hh / 32, 256 * 256, lane);
    int half = lane >> 4, nl = lane & 15;
#pragma unroll
    for (int r = 0; r < 8; ++r) gl[w][r + 8 * half][nl] = acc[r];
    __syncthreads();

    int b = threadIdx.x >> 4, jl = threadIdx.x & 15, j = j0 + jl;
    float gi = gl[0][b][jl] + xg[(t * 4096 + 0 * Hh + j) * Bb + b];
    float gf = gl[1][b][jl] + xg[(t * 4096 + 1 * Hh + j) * Bb + b];
    float gg = gl[2][b][jl] + xg[(t * 4096 + 2 * Hh + j) * Bb + b];
    float go = gl[3][b][jl] + xg[(t * 4096 + 3 * Hh + j) * Bb + b];
    float cp = (t > 0) ? cs[(b * Tt + (t - 1)) * Hh + j] : 0.f;
    float c = sigf(gf) * cp + sigf(gi) * tanhf(gg);
    float h = sigf(go) * tanhf(c);
    hs[(b * Tt + t) * Hh + j] = h;
    cs[(b * Tt + t) * Hh + j] = c;
    hsh[b][jl] = h;
    __syncthreads();
    if ((jl & 1) == 0) {                             // pack next-step A (pairs), zero pad rows
        int kp = j >> 1;
        Aout[kp * 16 + b]     = pack2bf(hsh[b][jl], hsh[b][jl + 1]);
        Aout[kp * 16 + 8 + b] = 0u;
    }
}

// ---------------- rank scores: relu(hs@Wr1^T)@Wr2^T ----------------
__global__ void rank_kernel(const float* __restrict__ hs, const float* __restrict__ Wr1,
                            const float* __restrict__ Wr2, float* __restrict__ scores) {
    __shared__ float red[128];
    int bt = blockIdx.x;
    int c = threadIdx.x;
    const float* hrow = hs + bt * Hh;
    const float* wrow = Wr1 + c * Hh;
    float acc = 0.f;
    for (int k = 0; k < Hh; ++k) acc += hrow[k] * wrow[k];
    acc = fmaxf(acc, 0.f) * Wr2[c];
    red[c] = acc; __syncthreads();
    for (int s = 64; s > 0; s >>= 1) { if (c < s) red[c] += red[c + s]; __syncthreads(); }
    if (c == 0) scores[bt] = red[0];
}

// ---------------- greedy tree build (post-order emit) ----------------
__global__ void tree_build(const float* __restrict__ scores, const int* __restrict__ length,
                           int* __restrict__ npos, int* __restrict__ nl,
                           int* __restrict__ nr, int* __restrict__ ncnt) {
    int b = blockIdx.x;
    if (threadIdx.x != 0) return;
    const float* sc = scores + b * Tt;
    int len = length[b];
    int st_s[80], st_e[80], st_ph[80], st_l[80], st_r[80], st_p[80];
    int sp = 0, cnt = 0, ret = -1;
    int* P = npos + b * Tt; int* L = nl + b * Tt; int* R = nr + b * Tt;
    if (len >= 2) {
        st_s[0] = 0; st_e[0] = len; st_ph[0] = 0; sp = 1;
        while (sp > 0) {
            int i = sp - 1;
            if (st_ph[i] == 0) {
                int s = st_s[i], e = st_e[i];
                int pos = s; float best = sc[s];
                for (int k = s + 1; k < e; ++k) if (sc[k] > best) { best = sc[k]; pos = k; }
                st_p[i] = pos; st_ph[i] = 1;
                int ls = s, le = pos;
                if (le <= ls)          st_l[i] = -1;
                else if (le == ls + 1) st_l[i] = LEAF_BASE + ls;
                else { st_l[i] = -2; st_s[sp] = ls; st_e[sp] = le; st_ph[sp] = 0; sp++; continue; }
            }
            if (st_ph[i] == 1) {
                if (st_l[i] == -2) st_l[i] = ret;
                st_ph[i] = 2;
                int rs = st_p[i] + 1, re = st_e[i];
                if (re <= rs)          st_r[i] = -1;
                else if (re == rs + 1) st_r[i] = LEAF_BASE + rs;
                else { st_r[i] = -2; st_s[sp] = rs; st_e[sp] = re; st_ph[sp] = 0; sp++; continue; }
            }
            if (st_r[i] == -2) st_r[i] = ret;
            int id = cnt++;
            P[id] = st_p[i]; L[id] = st_l[i]; R[id] = st_r[i];
            ret = id; sp--;
        }
    }
    ncnt[b] = cnt;
}

// ---------------- build cat(hl,hr,hm) A-fragments for node round k ----------------
__global__ void cat_prep(int k, const int* __restrict__ npos, const int* __restrict__ nl,
                         const int* __restrict__ nr, const int* __restrict__ ncnt,
                         const float* __restrict__ hs, const float* __restrict__ node_h,
                         unsigned int* __restrict__ catApk) {
    int idx = blockIdx.x * 256 + threadIdx.x;           // 1536 pairs x 16 rows
    if (idx >= 1536 * 16) return;
    int m = idx & 15, p = idx >> 4;
    if (m >= Bb) { catApk[p * 16 + m] = 0u; return; }
    int b = m;
    float v0 = 0.f, v1 = 0.f;
    if (k < ncnt[b]) {
        int j = 2 * p, seg = j >> 10, jj = j & 1023;
        int ref;
        if (seg == 0)      ref = nl[b * Tt + k];
        else if (seg == 1) ref = nr[b * Tt + k];
        else               ref = LEAF_BASE + npos[b * Tt + k];   // hm = hard select
        if (ref >= LEAF_BASE) {
            const float* src = hs + (b * Tt + (ref - LEAF_BASE)) * Hh;
            v0 = src[jj]; v1 = src[jj + 1];
        } else if (ref >= 0) {
            const float* src = node_h + (b * 64 + ref) * Hh;
            v0 = src[jj]; v1 = src[jj + 1];
        }
    }
    catApk[p * 16 + m] = pack2bf(v0, v1);
}

__device__ __forceinline__ float fetch_c(int ref, int b, int j,
                                         const float* __restrict__ cs,
                                         const float* __restrict__ node_c) {
    if (ref < 0) return 0.f;
    if (ref >= LEAF_BASE) return cs[(b * Tt + (ref - LEAF_BASE)) * Hh + j];
    return node_c[(b * 64 + ref) * Hh + j];
}

// ---------------- combine round k: M=8(pad16) x N=6144 x K=3072 ----------------
// grid 64 x 192 (6 waves). Block owns j0..j0+15; wave g = gate g (i,fl,fr,fm,u,o).
__global__ void combine_step(int k, const unsigned int* __restrict__ catApk,
                             const unsigned int* __restrict__ Wcs,
                             const float* __restrict__ bc,
                             const int* __restrict__ npos, const int* __restrict__ nl,
                             const int* __restrict__ nr, const int* __restrict__ ncnt,
                             const float* __restrict__ cs,
                             float* __restrict__ node_h, float* __restrict__ node_c) {
    __shared__ float gl[6][16][16];
    int lane = threadIdx.x & 31, w = threadIdx.x >> 5;
    int j0 = blockIdx.x * 16;
    int ntile = w * 64 + blockIdx.x;                    // n = w*1024 + j0 + nl in 6144
    v8f acc = gemm_tile(catApk, Wcs + ntile * 256, 3072 / 32, 384 * 256, lane);
    int half = lane >> 4, nlc = lane & 15;
#pragma unroll
    for (int r = 0; r < 8; ++r) gl[w][r + 8 * half][nlc] = acc[r];
    __syncthreads();

    int tid = threadIdx.x;
    if (tid < 128) {
        int b = tid >> 4, jl = tid & 15, j = j0 + jl;
        if (k < ncnt[b]) {
            float gi  = gl[0][b][jl] + bc[0 * Hh + j];
            float gfl = gl[1][b][jl] + bc[1 * Hh + j];
            float gfr = gl[2][b][jl] + bc[2 * Hh + j];
            float gfm = gl[3][b][jl] + bc[3 * Hh + j];
            float gu  = gl[4][b][jl] + bc[4 * Hh + j];
            float go  = gl[5][b][jl] + bc[5 * Hh + j];
            int lref = nl[b * Tt + k], rref = nr[b * Tt + k], pos = npos[b * Tt + k];
            float cl = fetch_c(lref, b, j, cs, node_c);
            float cr = fetch_c(rref, b, j, cs, node_c);
            float cm = cs[(b * Tt + pos) * Hh + j];
            float c = sigf(gfl + 1.f) * cl + sigf(gfr + 1.f) * cr
                    + sigf(gfm + 1.f) * cm + sigf(gi) * tanhf(gu);
            float h = sigf(go) * tanhf(c);
            node_h[(b * 64 + k) * Hh + j] = h;
            node_c[(b * 64 + k) * Hh + j] = c;
        }
    }
}

// ---------------- gather roots ----------------
__global__ void finalize(const int* __restrict__ ncnt, const float* __restrict__ node_h,
                         const float* __restrict__ node_c, const float* __restrict__ hs,
                         const float* __restrict__ cs, float* __restrict__ out) {
    int b = blockIdx.x;
    int j = blockIdx.y * 256 + threadIdx.x;
    int n = ncnt[b];
    float h, c;
    if (n > 0) { h = node_h[(b * 64 + n - 1) * Hh + j]; c = node_c[(b * 64 + n - 1) * Hh + j]; }
    else       { h = hs[(b * Tt) * Hh + j];             c = cs[(b * Tt) * Hh + j]; }
    out[b * Hh + j]              = h;
    out[Bb * Hh + b * Hh + j]    = c;
}

// ---------------- host launch ----------------
static inline size_t alignup(size_t x) { return (x + 255) & ~size_t(255); }

extern "C" void kernel_launch(void* const* d_in, const int* in_sizes, int n_in,
                              void* d_out, int out_size, void* d_ws, size_t ws_size,
                              hipStream_t stream) {
    const float* emb  = (const float*)d_in[0];
    const int*   len  = (const int*)d_in[2];
    const float* W_ih = (const float*)d_in[3];
    const float* W_hh = (const float*)d_in[4];
    const float* b_ih = (const float*)d_in[5];
    const float* b_hh = (const float*)d_in[6];
    const float* Wr1  = (const float*)d_in[7];
    const float* Wr2  = (const float*)d_in[8];
    const float* Wc   = (const float*)d_in[9];
    const float* bc   = (const float*)d_in[10];
    float* out = (float*)d_out;

    char* ws = (char*)d_ws;
    size_t off = 0;
    auto take = [&](size_t bytes) { char* p = ws + off; off = alignup(off + bytes); return p; };

    unsigned int* WihS   = (unsigned int*)take(4096u * 512u / 2u * 4u);     // 4 MB
    unsigned int* WhhS   = (unsigned int*)take(4096u * 1024u / 2u * 4u);    // 8 MB
    unsigned int* WcS    = (unsigned int*)take(6144u * 3072u / 2u * 4u);    // 37.7 MB
    unsigned int* embApk = (unsigned int*)take(32u * 256u * 16u * 4u);      // 512 KB
    float*        xg     = (float*)take((size_t)Tt * 4096u * Bb * 4u);      // 8 MB
    unsigned int* Aping  = (unsigned int*)take(512u * 16u * 4u);
    unsigned int* Apong  = (unsigned int*)take(512u * 16u * 4u);
    float*        hs     = (float*)take((size_t)Bb * Tt * Hh * 4u);         // 2 MB
    float*        cs     = (float*)take((size_t)Bb * Tt * Hh * 4u);         // 2 MB
    float*        scores = (float*)take((size_t)Bb * Tt * 4u);
    int*          npos   = (int*)take((size_t)Bb * Tt * 4u);
    int*          nlv    = (int*)take((size_t)Bb * Tt * 4u);
    int*          nrv    = (int*)take((size_t)Bb * Tt * 4u);
    int*          ncnt   = (int*)take(32u);
    float*        node_h = (float*)take((size_t)Bb * 64u * Hh * 4u);        // 2 MB
    float*        node_c = (float*)take((size_t)Bb * 64u * Hh * 4u);        // 2 MB
    unsigned int* catApk = (unsigned int*)take(1536u * 16u * 4u);
    (void)ws_size; (void)in_sizes; (void)n_in; (void)out_size;

    // weight swizzles (bf16 WMMA fragment layout), done fresh each call (deterministic)
    swizzleB<<<(4096 * 512 / 2 + 255) / 256, 256, 0, stream>>>(W_ih, WihS, 4096, 512);
    swizzleB<<<(4096 * 1024 / 2 + 255) / 256, 256, 0, stream>>>(W_hh, WhhS, 4096, 1024);
    swizzleB<<<(6144 * 3072 / 2 + 255) / 256, 256, 0, stream>>>(Wc, WcS, 6144, 3072);
    packA_emb<<<(32 * 256 * 16 + 255) / 256, 256, 0, stream>>>(emb, embApk);

    // input-gate precompute (parallel over all T)
    gemm_xg<<<2048, 128, 0, stream>>>(embApk, WihS, b_ih, b_hh, xg);

    // LSTM recurrence: 64 ordered step kernels, ping-pong packed h
    zero_u32<<<(512 * 16 + 255) / 256, 256, 0, stream>>>(Aping, 512 * 16);
    for (int t = 0; t < Tt; ++t) {
        const unsigned int* Ain = (t & 1) ? Apong : Aping;
        unsigned int* Aout      = (t & 1) ? Aping : Apong;
        lstm_step<<<64, 128, 0, stream>>>(t, Ain, Aout, WhhS, xg, hs, cs);
    }

    rank_kernel<<<Bb * Tt, 128, 0, stream>>>(hs, Wr1, Wr2, scores);
    tree_build<<<Bb, 32, 0, stream>>>(scores, len, npos, nlv, nrv, ncnt);

    // tree combines: post-order rounds, all 8 batches per round as one M=8 GEMM
    for (int k = 0; k < Tt - 1; ++k) {
        cat_prep<<<(1536 * 16 + 255) / 256, 256, 0, stream>>>(k, npos, nlv, nrv, ncnt,
                                                              hs, node_h, catApk);
        combine_step<<<64, 192, 0, stream>>>(k, catApk, WcS, bc, npos, nlv, nrv, ncnt,
                                             cs, node_h, node_c);
    }

    finalize<<<dim3(Bb, 4), 256, 0, stream>>>(ncnt, node_h, node_c, hs, cs, out);
}
